// BayesianNet_79250736546052
// MI455X (gfx1250) — matrix operand order repaired and checked by reading, not compiled, not run on hardware
//
#include <hip/hip_runtime.h>
#include <hip/hip_bf16.h>

// ---------------- types / helpers ----------------
typedef _Float16 v16h __attribute__((ext_vector_type(16)));
typedef _Float16 v8h  __attribute__((ext_vector_type(8)));
typedef float    v8f  __attribute__((ext_vector_type(8)));

__device__ __forceinline__ v8f zero8() {
    v8f z;
    #pragma unroll
    for (int i = 0; i < 8; ++i) z[i] = 0.f;
    return z;
}

// A fragment (16x32 f16, M x K): lane l<16 holds row M=l with K = 0..15;
// lane l>=16 holds row M=l-16 with K = 16..31.  kb0 = (l<16)?0:16.
__device__ __forceinline__ v16h load_frag_a(const _Float16* row, int kb0) {
    v8h lo = *(const v8h*)(row + kb0);
    v8h hi = *(const v8h*)(row + kb0 + 8);
    return __builtin_shufflevector(lo, hi, 0,1,2,3,4,5,6,7,8,9,10,11,12,13,14,15);
}
// B fragment (32x16 f16, K x N): lane l<16 holds col N=l with K=0..15;
// lane l>=16 holds col N=l-16 with K=16..31.  kbB = (l<16)?0:16.
__device__ __forceinline__ v16h load_frag_b(const _Float16* col, int kbB) {
    v8h lo = *(const v8h*)(col + kbB);
    v8h hi = *(const v8h*)(col + kbB + 8);
    return __builtin_shufflevector(lo, hi, 0,1,2,3,4,5,6,7,8,9,10,11,12,13,14,15);
}
__device__ __forceinline__ v8f wmma_f16(v16h a, v16h b, v8f c) {
    return __builtin_amdgcn_wmma_f32_16x16x32_f16(false, a, false, b, (short)0, c, false, false);
}
__device__ __forceinline__ float sigmoidf_(float x) { return 1.f / (1.f + expf(-x)); }

// ---------------- model constants ----------------
#define BSZ   128
#define TSEQ  256
#define IDIM_ 64
#define HDIM_ 128
#define OC1_  100
#define ANF_  64
#define S_    3364   // 58*58
#define SP_   3392   // padded to mult of 32
#define F_    3620
#define FP_   3648
#define HID_  1810
#define HIDP_ 1824

// ---------------- pack kernels ----------------
__global__ void pack_f16(const float* __restrict__ src, _Float16* __restrict__ dst,
                         int R, int C, int srcStride, int srcColOff, int Rp, int Cp) {
    long idx = (long)blockIdx.x * blockDim.x + threadIdx.x;
    long tot = (long)Rp * Cp;
    if (idx >= tot) return;
    int c = (int)(idx % Cp), r = (int)(idx / Cp);
    float v = (r < R && c < C) ? src[(size_t)r * srcStride + srcColOff + c] : 0.f;
    dst[idx] = (_Float16)v;
}

// conv2 weights [100][100][3][3] -> f16 [9][112 oc][128 ic] zero padded
__global__ void pack_convw(const float* __restrict__ w, _Float16* __restrict__ dst) {
    int idx = blockIdx.x * blockDim.x + threadIdx.x;
    if (idx >= 9 * 112 * 128) return;
    int ic = idx & 127; int t = idx >> 7;
    int oc = t % 112;   int kk = t / 112;
    int ky = kk / 3, kx = kk % 3;
    float v = (oc < OC1_ && ic < OC1_) ? w[(((size_t)oc * OC1_ + ic) * 3 + ky) * 3 + kx] : 0.f;
    dst[idx] = (_Float16)v;
}

__global__ void bias_sum(const float* a, const float* b, float* o, int n) {
    int i = blockIdx.x * blockDim.x + threadIdx.x;
    if (i < n) o[i] = a[i] + b[i];
}

// ---------------- conv1 direct (K=27, tiny) ----------------
__global__ void conv1_direct(const float* __restrict__ x, const float* __restrict__ w,
                             const float* __restrict__ bias, _Float16* __restrict__ out) {
    long idx = (long)blockIdx.x * blockDim.x + threadIdx.x;
    long tot = (long)BSZ * OC1_ * 62 * 62;
    if (idx >= tot) return;
    int xx = (int)(idx % 62); long t = idx / 62;
    int yy = (int)(t % 62); t /= 62;
    int oc = (int)(t % OC1_); int b = (int)(t / OC1_);
    float acc = bias[oc];
    #pragma unroll
    for (int ic = 0; ic < 3; ++ic)
        #pragma unroll
        for (int ky = 0; ky < 3; ++ky)
            #pragma unroll
            for (int kx = 0; kx < 3; ++kx)
                acc += x[(((size_t)b * 3 + ic) * 64 + yy + ky) * 64 + xx + kx] *
                       w[(((size_t)oc * 3 + ic) * 3 + ky) * 3 + kx];
    out[idx] = (_Float16)(acc > 0.f ? acc : 0.f);
}

// ---------------- implicit-GEMM conv (100->100, 3x3 VALID) via WMMA ----------------
// One wave computes ALL 7 oc-tiles (112 padded ocs) for one 16-pixel tile, so each
// gathered B fragment (input activations, strided over channels) is reused 7x.
// out[b][oc][p] = relu( sum_{ky,kx,ic} wpack[kk][oc][ic] * in[b][ic][y+ky][x+kx] + bias[oc] )
__global__ __launch_bounds__(256)
void conv_wmma(const _Float16* __restrict__ in, const _Float16* __restrict__ wpack,
               const float* __restrict__ bias, _Float16* __restrict__ out,
               int Hin, int Win, int Wout, int P, int ldout) {
    int HWin = Hin * Win;
    int Ntiles = (P + 15) >> 4;
    int wave = blockIdx.x * (blockDim.x >> 5) + (threadIdx.x >> 5);
    int lane = threadIdx.x & 31;
    if (wave >= BSZ * Ntiles) return;
    int b = wave / Ntiles, ptile = wave % Ntiles;
    int n = lane & 15;
    int p = ptile * 16 + n;
    bool pok = p < P;
    int pc = pok ? p : (P - 1);
    int y = pc / Wout, x = pc - y * Wout;
    const _Float16* inb = in + (size_t)b * OC1_ * HWin;
    int kbA = (lane < 16) ? 0 : 16;
    int kbB = (lane < 16) ? 0 : 16;
    v8f acc[7];
    #pragma unroll
    for (int mt = 0; mt < 7; ++mt) acc[mt] = zero8();
    for (int kk = 0; kk < 9; ++kk) {
        int ky = kk / 3, kx = kk - ky * 3;
        const _Float16* ib = inb + (size_t)(y + ky) * Win + (x + kx);
        const _Float16* Abase = wpack + (size_t)kk * 112 * 128 + (size_t)n * 128;
        #pragma unroll
        for (int kc = 0; kc < 4; ++kc) {
            // gather B fragment once (16 strided f16 loads over input channels)
            v16h bf;
            int icb = kc * 32 + kbB;
            #pragma unroll
            for (int e = 0; e < 16; ++e) {
                int ic = icb + e;
                bf[e] = (ic < OC1_) ? ib[(size_t)ic * HWin] : (_Float16)0;
            }
            // reuse it for all 7 oc tiles
            #pragma unroll
            for (int mt = 0; mt < 7; ++mt) {
                v16h a = load_frag_a(Abase + (size_t)mt * 16 * 128 + kc * 32, kbA);
                acc[mt] = wmma_f16(a, bf, acc[mt]);
            }
        }
    }
    int moff = (lane >> 4) << 3;
    #pragma unroll
    for (int mt = 0; mt < 7; ++mt) {
        #pragma unroll
        for (int r = 0; r < 8; ++r) {
            int oc = mt * 16 + moff + r;
            if (oc < OC1_ && pok) {
                float v = acc[mt][r] + bias[oc];
                if (v < 0.f) v = 0.f;
                out[((size_t)b * OC1_ + oc) * ldout + (ptile * 16 + n)] = (_Float16)v;
            }
        }
    }
}

// ---------------- generic WMMA GEMM:  C[j][m] = A[m][:] . Bsrc[row(j)][:] (+bias[m]) ----------------
__global__ __launch_bounds__(256)
void gemm_wmma(const _Float16* __restrict__ A, const _Float16* __restrict__ Bsrc,
               float* __restrict__ C, const float* __restrict__ bias,
               int Mtiles, int M, int N, int Kchunks,
               int lda, int ldb, int ldc, int jmap, int relu) {
    int wave = blockIdx.x * (blockDim.x >> 5) + (threadIdx.x >> 5);
    int lane = threadIdx.x & 31;
    int Ntiles = (N + 15) >> 4;
    if (wave >= Mtiles * Ntiles) return;
    int mtile = wave / Ntiles, ntile = wave % Ntiles;
    int n = lane & 15;
    int jreal = ntile * 16 + n;
    int j = (jreal < N) ? jreal : (N - 1);
    int row = j;
    if (jmap == 1) { int bb = j & 127; int tt = j >> 7; row = bb * TSEQ + tt; }
    const _Float16* Arow = A + (size_t)(mtile * 16 + n) * lda;
    const _Float16* Brow = Bsrc + (size_t)row * ldb;
    int kbA = (lane < 16) ? 0 : 16;
    int kbB = (lane < 16) ? 0 : 16;
    v8f acc = zero8();
    for (int kc = 0; kc < Kchunks; ++kc) {
        v16h a  = load_frag_a(Arow + (size_t)kc * 32, kbA);
        v16h bf = load_frag_b(Brow + (size_t)kc * 32, kbB);
        acc = wmma_f16(a, bf, acc);
    }
    if (jreal < N) {
        int moff = (lane >> 4) << 3;
        float* Crow = C + (size_t)jreal * ldc;
        #pragma unroll
        for (int r = 0; r < 8; ++r) {
            int m = mtile * 16 + moff + r;
            if (m < M) {
                float v = acc[r] + (bias ? bias[m] : 0.f);
                if (relu && v < 0.f) v = 0.f;
                Crow[m] = v;
            }
        }
    }
}

// ---------------- LSTM recurrence: one persistent workgroup, h in LDS, c in regs ----------------
// xg: [T][B][512] f32 (x@Wih^T + b_ih + b_hh), whh: f16 [512][128]
__global__ __launch_bounds__(1024)
void lstm_rec(const float* __restrict__ xg, const _Float16* __restrict__ whh,
              _Float16* __restrict__ yout, float* __restrict__ hn, int hnOff) {
    __shared__ _Float16 hbuf[HDIM_ * HDIM_];   // [batch 128][hidden 128], 32 KB
    int tid = threadIdx.x;
    int wave = tid >> 5, lane = tid & 31;
    int mtile = wave >> 2;   // batch tile 0..7
    int q = wave & 3;        // column group
    int n = lane & 15;
    int kbA = (lane < 16) ? 0 : 16;
    int kbB = (lane < 16) ? 0 : 16;
    int moff = (lane >> 4) << 3;
    for (int i = tid; i < HDIM_ * HDIM_; i += 1024) hbuf[i] = (_Float16)0;
    float creg[2][8];
    #pragma unroll
    for (int j = 0; j < 2; ++j)
        #pragma unroll
        for (int r = 0; r < 8; ++r) creg[j][r] = 0.f;
    __syncthreads();

    for (int t = 0; t < TSEQ; ++t) {
        v16h afr[4];
        const _Float16* hrow = hbuf + (size_t)(mtile * 16 + n) * HDIM_;
        #pragma unroll
        for (int kc = 0; kc < 4; ++kc) afr[kc] = load_frag_a(hrow + kc * 32, kbA);
        __syncthreads();   // everyone has read h_t before it gets overwritten

        const float* xgb = xg + (size_t)t * BSZ * 512;
        #pragma unroll
        for (int j = 0; j < 2; ++j) {
            int hc = (q * 2 + j) * 16 + n;   // hidden column 0..127
            v8f acc[4];
            #pragma unroll
            for (int g = 0; g < 4; ++g) {
                #pragma unroll
                for (int r = 0; r < 8; ++r) {
                    int batch = mtile * 16 + moff + r;
                    acc[g][r] = xgb[(size_t)batch * 512 + g * HDIM_ + hc];
                }
            }
            #pragma unroll
            for (int g = 0; g < 4; ++g) {
                const _Float16* wrow = whh + (size_t)(g * HDIM_ + hc) * HDIM_;
                #pragma unroll
                for (int kc = 0; kc < 4; ++kc) {
                    v16h bf = load_frag_b(wrow + kc * 32, kbB);
                    acc[g] = wmma_f16(afr[kc], bf, acc[g]);
                }
            }
            #pragma unroll
            for (int r = 0; r < 8; ++r) {
                float iv = sigmoidf_(acc[0][r]);
                float fv = sigmoidf_(acc[1][r]);
                float gv = tanhf(acc[2][r]);
                float ov = sigmoidf_(acc[3][r]);
                float c = fv * creg[j][r] + iv * gv;
                creg[j][r] = c;
                float h = ov * tanhf(c);
                int batch = mtile * 16 + moff + r;
                hbuf[(size_t)batch * HDIM_ + hc] = (_Float16)h;
                if (yout) yout[((size_t)t * BSZ + batch) * HDIM_ + hc] = (_Float16)h;
                if (t == TSEQ - 1) hn[(size_t)batch * 256 + hnOff + hc] = h;
            }
        }
        __syncthreads();   // h_{t+1} visible to all waves
    }
}

// ---------------- attention: hidden-state contribution ----------------
__global__ void attn_hvec(const float* __restrict__ hn, const float* __restrict__ attn1_w,
                          float* __restrict__ hvec) {
    int idx = blockIdx.x * blockDim.x + threadIdx.x;
    if (idx >= BSZ * ANF_) return;
    int a = idx & 63, b = idx >> 6;
    float s = 0.f;
    for (int j = 0; j < 256; ++j)
        s += hn[(size_t)b * 256 + j] * attn1_w[(size_t)a * F_ + S_ + j];
    hvec[idx] = s;
}

// ---------------- attention: scores + softmax over channels ----------------
__global__ void attn_score(const float* __restrict__ A1, const float* __restrict__ hvec,
                           const float* __restrict__ attn1_b, const float* __restrict__ attn2_w,
                           const float* __restrict__ attn2_b, float* __restrict__ aw) {
    __shared__ float sc[128];
    int b = blockIdx.x, c = threadIdx.x;
    float s = -1e30f;
    if (c < OC1_) {
        float acc = attn2_b[0];
        const float* row = A1 + (size_t)(b * OC1_ + c) * ANF_;
        for (int a = 0; a < ANF_; ++a)
            acc += tanhf(row[a] + hvec[b * ANF_ + a] + attn1_b[a]) * attn2_w[a];
        s = acc;
    }
    sc[c] = s;
    __syncthreads();
    if (c < OC1_) {
        float mx = -1e30f;
        for (int i = 0; i < OC1_; ++i) mx = fmaxf(mx, sc[i]);
        float sum = 0.f;
        for (int i = 0; i < OC1_; ++i) sum += expf(sc[i] - mx);
        aw[b * OC1_ + c] = expf(sc[c] - mx) / sum;
    }
}

// ---------------- attention context -> mh[b][0:S] (f16) ----------------
__global__ void attn_ctx(const _Float16* __restrict__ xd, const float* __restrict__ aw,
                         _Float16* __restrict__ mh) {
    __shared__ float a_s[OC1_];
    int b = blockIdx.y;
    int s0 = blockIdx.x * 256 + threadIdx.x;
    if (threadIdx.x < OC1_) a_s[threadIdx.x] = aw[b * OC1_ + threadIdx.x];
    __syncthreads();
    if (s0 >= S_) return;
    float acc = 0.f;
    const _Float16* xb = xd + (size_t)b * OC1_ * SP_ + s0;
    for (int c = 0; c < OC1_; ++c) acc += (float)xb[(size_t)c * SP_] * a_s[c];
    mh[(size_t)b * FP_ + s0] = (_Float16)acc;
}

// ---------------- mh tail: hn (f32->f16) + zero pad ----------------
__global__ void fill_mtail(const float* __restrict__ hn, _Float16* __restrict__ mh) {
    int idx = blockIdx.x * blockDim.x + threadIdx.x;
    const int tail = FP_ - S_; // 284
    if (idx >= BSZ * tail) return;
    int c = idx % tail, b = idx / tail;
    float v = (c < 256) ? hn[(size_t)b * 256 + c] : 0.f;
    mh[(size_t)b * FP_ + S_ + c] = (_Float16)v;
}

// ---------------- fc2 reduction ----------------
__global__ void fc2_out(const float* __restrict__ h1, const float* __restrict__ fc2_w,
                        const float* __restrict__ fc2_b, float* __restrict__ out) {
    __shared__ float red[256];
    int b = blockIdx.x, t = threadIdx.x;
    float s = 0.f;
    for (int i = t; i < HID_; i += 256) s += h1[(size_t)b * HIDP_ + i] * fc2_w[i];
    red[t] = s;
    __syncthreads();
    for (int o = 128; o > 0; o >>= 1) {
        if (t < o) red[t] += red[t + o];
        __syncthreads();
    }
    if (t == 0) out[b] = red[0] + fc2_b[0];
}

// ---------------- host orchestration ----------------
extern "C" void kernel_launch(void* const* d_in, const int* in_sizes, int n_in,
                              void* d_out, int out_size, void* d_ws, size_t ws_size,
                              hipStream_t stream) {
    const float* x1      = (const float*)d_in[0];
    const float* x2      = (const float*)d_in[1];
    const float* conv1_w = (const float*)d_in[2];
    const float* conv1_b = (const float*)d_in[3];
    const float* conv2a_w= (const float*)d_in[4];
    const float* conv2a_b= (const float*)d_in[5];
    const float* conv2b_w= (const float*)d_in[6];
    const float* conv2b_b= (const float*)d_in[7];
    const float* w_ih0   = (const float*)d_in[8];
    const float* w_hh0   = (const float*)d_in[9];
    const float* b_ih0   = (const float*)d_in[10];
    const float* b_hh0   = (const float*)d_in[11];
    const float* w_ih1   = (const float*)d_in[12];
    const float* w_hh1   = (const float*)d_in[13];
    const float* b_ih1   = (const float*)d_in[14];
    const float* b_hh1   = (const float*)d_in[15];
    const float* attn1_w = (const float*)d_in[16];
    const float* attn1_b = (const float*)d_in[17];
    const float* attn2_w = (const float*)d_in[18];
    const float* attn2_b = (const float*)d_in[19];
    const float* fc1_w   = (const float*)d_in[20];
    const float* fc1_b   = (const float*)d_in[21];
    const float* fc2_w   = (const float*)d_in[22];
    const float* fc2_b   = (const float*)d_in[23];
    float* out = (float*)d_out;

    size_t off = 0;
    auto alloc = [&](size_t bytes) -> void* {
        void* p = (char*)d_ws + off;
        off = (off + bytes + 255) & ~(size_t)255;
        return p;
    };
    _Float16* act1   = (_Float16*)alloc((size_t)BSZ * OC1_ * 62 * 62 * 2); // 98.4 MB
    _Float16* act2   = (_Float16*)alloc((size_t)BSZ * OC1_ * 60 * 60 * 2); // 92.2 MB
    _Float16* xd     = (_Float16*)alloc((size_t)BSZ * OC1_ * SP_ * 2);     // 86.8 MB
    _Float16* wpackA = (_Float16*)alloc((size_t)9 * 112 * 128 * 2);
    _Float16* wpackB = (_Float16*)alloc((size_t)9 * 112 * 128 * 2);
    _Float16* wih0h  = (_Float16*)alloc((size_t)512 * 64 * 2);
    _Float16* whh0h  = (_Float16*)alloc((size_t)512 * 128 * 2);
    _Float16* wih1h  = (_Float16*)alloc((size_t)512 * 128 * 2);
    _Float16* whh1h  = (_Float16*)alloc((size_t)512 * 128 * 2);
    _Float16* x2h    = (_Float16*)alloc((size_t)BSZ * TSEQ * 64 * 2);
    float*    xg0    = (float*)   alloc((size_t)TSEQ * BSZ * 512 * 4);     // 67.1 MB
    float*    biasg0 = (float*)   alloc(512 * 4);
    float*    biasg1 = (float*)   alloc(512 * 4);
    float*    hn     = (float*)   alloc((size_t)BSZ * 256 * 4);
    _Float16* attn1sh= (_Float16*)alloc((size_t)ANF_ * SP_ * 2);
    float*    A1     = (float*)   alloc((size_t)BSZ * OC1_ * ANF_ * 4);
    float*    hvec   = (float*)   alloc((size_t)BSZ * ANF_ * 4);
    float*    aw     = (float*)   alloc((size_t)BSZ * OC1_ * 4);
    _Float16* mh     = (_Float16*)alloc((size_t)BSZ * FP_ * 2);
    _Float16* fc1h   = (_Float16*)alloc((size_t)HIDP_ * FP_ * 2);          // 13.3 MB
    float*    h1     = (float*)   alloc((size_t)BSZ * HIDP_ * 4);
    // aliases (stream order makes these safe): xg1 reuses act1, y0 reuses act2
    float*    xg1 = (float*)act1;      // needs 67.1 MB <= 98.4 MB
    _Float16* y0  = (_Float16*)act2;   // needs  8.4 MB <= 92.2 MB
    (void)ws_size; (void)n_in; (void)in_sizes; (void)out_size;

    auto nblk = [](long n, int b) { return (unsigned)((n + b - 1) / b); };

    // ---- packs ----
    pack_convw<<<nblk(9 * 112 * 128, 256), 256, 0, stream>>>(conv2a_w, wpackA);
    pack_convw<<<nblk(9 * 112 * 128, 256), 256, 0, stream>>>(conv2b_w, wpackB);
    pack_f16<<<nblk((long)512 * 64, 256), 256, 0, stream>>>(w_ih0, wih0h, 512, 64, 64, 0, 512, 64);
    pack_f16<<<nblk((long)512 * 128, 256), 256, 0, stream>>>(w_hh0, whh0h, 512, 128, 128, 0, 512, 128);
    pack_f16<<<nblk((long)512 * 128, 256), 256, 0, stream>>>(w_ih1, wih1h, 512, 128, 128, 0, 512, 128);
    pack_f16<<<nblk((long)512 * 128, 256), 256, 0, stream>>>(w_hh1, whh1h, 512, 128, 128, 0, 512, 128);
    pack_f16<<<nblk((long)BSZ * TSEQ * 64, 256), 256, 0, stream>>>(x2, x2h, BSZ * TSEQ, 64, 64, 0, BSZ * TSEQ, 64);
    pack_f16<<<nblk((long)ANF_ * SP_, 256), 256, 0, stream>>>(attn1_w, attn1sh, ANF_, S_, F_, 0, ANF_, SP_);
    pack_f16<<<nblk((long)HIDP_ * FP_, 256), 256, 0, stream>>>(fc1_w, fc1h, HID_, F_, F_, 0, HIDP_, FP_);
    bias_sum<<<2, 256, 0, stream>>>(b_ih0, b_hh0, biasg0, 512);
    bias_sum<<<2, 256, 0, stream>>>(b_ih1, b_hh1, biasg1, 512);

    // ---- CNN branch ----
    conv1_direct<<<nblk((long)BSZ * OC1_ * 62 * 62, 256), 256, 0, stream>>>(x1, conv1_w, conv1_b, act1);
    {   // conv2a: 62x62 -> 60x60, P=3600; one wave per 16-pixel tile (all 7 oc tiles)
        int P = 3600, Ntiles = (P + 15) >> 4;
        long tiles = (long)BSZ * Ntiles;
        conv_wmma<<<nblk(tiles, 8), 256, 0, stream>>>(act1, wpackA, conv2a_b, act2, 62, 62, 60, P, 3600);
    }
    (void)hipMemsetAsync(xd, 0, (size_t)BSZ * OC1_ * SP_ * 2, stream); // zero K-pad for attention GEMM
    {   // conv2b: 60x60 -> 58x58, P=3364, row stride SP_
        int P = S_, Ntiles = (P + 15) >> 4;
        long tiles = (long)BSZ * Ntiles;
        conv_wmma<<<nblk(tiles, 8), 256, 0, stream>>>(act2, wpackB, conv2b_b, xd, 60, 60, 58, P, SP_);
    }

    // ---- LSTM branch ----
    {   // xg0[t][b][512] = x2 @ w_ih0^T + biases  (jmap=1: j=t*B+b -> row b*T+t)
        int N = BSZ * TSEQ, Ntiles = (N + 15) >> 4;
        gemm_wmma<<<nblk((long)32 * Ntiles, 8), 256, 0, stream>>>(
            wih0h, x2h, xg0, biasg0, 32, 512, N, 2, 64, 64, 512, 1, 0);
    }
    lstm_rec<<<1, 1024, 0, stream>>>(xg0, whh0h, y0, hn, 0);
    {   // xg1[t][b][512] = y0 @ w_ih1^T + biases  (identity j map)
        int N = BSZ * TSEQ, Ntiles = (N + 15) >> 4;
        gemm_wmma<<<nblk((long)32 * Ntiles, 8), 256, 0, stream>>>(
            wih1h, y0, xg1, biasg1, 32, 512, N, 4, 128, 128, 512, 0, 0);
    }
    lstm_rec<<<1, 1024, 0, stream>>>(xg1, whh1h, nullptr, hn, HDIM_);

    // ---- attention ----
    {   // A1[j=b*100+c][anf] = xd[j][:] @ attn1_w_s^T
        int N = BSZ * OC1_, Ntiles = (N + 15) >> 4;
        gemm_wmma<<<nblk((long)4 * Ntiles, 8), 256, 0, stream>>>(
            attn1sh, xd, A1, nullptr, 4, ANF_, N, SP_ / 32, SP_, SP_, ANF_, 0, 0);
    }
    attn_hvec<<<nblk((long)BSZ * ANF_, 256), 256, 0, stream>>>(hn, attn1_w, hvec);
    attn_score<<<BSZ, 128, 0, stream>>>(A1, hvec, attn1_b, attn2_w, attn2_b, aw);
    attn_ctx<<<dim3(nblk(S_, 256), BSZ), 256, 0, stream>>>(xd, aw, mh);
    fill_mtail<<<nblk((long)BSZ * (FP_ - S_), 256), 256, 0, stream>>>(hn, mh);

    // ---- fusion MLP ----
    {   // h1[b][m] = relu(mh[b] @ fc1_w^T + fc1_b)
        int Ntiles = (BSZ + 15) >> 4; // 8
        gemm_wmma<<<nblk((long)(HIDP_ / 16) * Ntiles, 8), 256, 0, stream>>>(
            fc1h, mh, h1, fc1_b, HIDP_ / 16, HID_, BSZ, FP_ / 32, FP_, FP_, HIDP_, 0, 1);
    }
    fc2_out<<<BSZ, 256, 0, stream>>>(h1, fc2_w, fc2_b, out);
}
